// MultiHeadAttentionLayer_32091995636370
// MI455X (gfx1250) — compile-verified, matrix operand order checked by reading, not run on hardware
//
#include <hip/hip_runtime.h>
#include <hip/hip_bf16.h>

// ---------------------------------------------------------------------------
// Problem constants (from reference): N=4, L=4096, HID=EMB=1024, NH=16, HD=64
// ---------------------------------------------------------------------------
#define NTOK   (4 * 4096)    // flattened tokens
#define F      1024          // feature dim (HID == EMB)
#define NHEAD  16
#define HDIM   64
#define TM     32            // tokens per workgroup (2 WMMA M-subtiles)
#define THREADS 512          // 16 waves (wave32)
#define NW     64            // output columns per wave (4 WMMA N-tiles)

typedef __attribute__((ext_vector_type(16))) __bf16 v16bf;
typedef __attribute__((ext_vector_type(8)))  float  v8f;

__device__ __forceinline__ unsigned short f2bf(float f) {
  unsigned u = __float_as_uint(f);
  u += 0x7FFFu + ((u >> 16) & 1u);          // round-to-nearest-even
  return (unsigned short)(u >> 16);
}
__device__ __forceinline__ float bf2f(unsigned short h) {
  return __uint_as_float(((unsigned)h) << 16);
}

// ---------------------------------------------------------------------------
// Kernel 1: fp32 -> bf16 weight conversion into workspace (L2-resident after)
// ---------------------------------------------------------------------------
__global__ void cvt_w_kernel(const float* __restrict__ src,
                             unsigned short* __restrict__ dst, int n) {
  int i = blockIdx.x * blockDim.x + threadIdx.x;   // one float4 per thread
  if (i * 4 < n) {
    float4 f = ((const float4*)src)[i];
    unsigned u0 = (unsigned)f2bf(f.x) | ((unsigned)f2bf(f.y) << 16);
    unsigned u1 = (unsigned)f2bf(f.z) | ((unsigned)f2bf(f.w) << 16);
    ((uint2*)dst)[i] = make_uint2(u0, u1);
  }
}

// ---------------------------------------------------------------------------
// Kernel 2: fully fused MHA for a 32-token tile.
//   LDS: xs (input tile / ctx), qs, ks, vs -- each [32][1024] bf16 = 64 KB
//   Total dynamic LDS: 256 KB (<= 320 KB WGP limit, one workgroup per WGP)
// ---------------------------------------------------------------------------
__global__ __launch_bounds__(THREADS)
void mha_fused_kernel(const float* __restrict__ Qin,
                      const float* __restrict__ Kin,
                      const float* __restrict__ Vin,
                      const unsigned short* __restrict__ Wqb, const float* __restrict__ bq,
                      const unsigned short* __restrict__ Wkb, const float* __restrict__ bk,
                      const unsigned short* __restrict__ Wvb, const float* __restrict__ bv,
                      const unsigned short* __restrict__ Wob, const float* __restrict__ bo,
                      float* __restrict__ out) {
  extern __shared__ unsigned short smem[];
  unsigned short* xs = smem;                 // input tile; later reused as ctx
  unsigned short* qs = smem + 1 * TM * F;
  unsigned short* ks = smem + 2 * TM * F;
  unsigned short* vs = smem + 3 * TM * F;

  const int t0   = blockIdx.x * TM;
  const int tid  = threadIdx.x;
  const int wave = tid >> 5;
  const int lane = tid & 31;
  const int mcol = lane & 15;     // A row / B column / C column owned by lane
  const int kh   = lane >> 4;     // lane-half selects K sub-run (and C row-half)

  // ---- stage a 32x1024 fp32 tile into LDS as bf16 ------------------------
  auto load_tile = [&](const float* __restrict__ src) {
    const float4* s = (const float4*)(src + (size_t)t0 * F);
    for (int i = tid; i < TM * F / 4; i += THREADS) {
      float4 f = s[i];
      unsigned u0 = (unsigned)f2bf(f.x) | ((unsigned)f2bf(f.y) << 16);
      unsigned u1 = (unsigned)f2bf(f.z) | ((unsigned)f2bf(f.w) << 16);
      ((uint2*)xs)[i] = make_uint2(u0, u1);
    }
  };

  // ---- [32 x 1024] = xs(bf16) @ W^T + bias, via v_wmma_f32_16x16x32_bf16 -
  // Wave w covers columns [w*64, w*64+64) as 4 N-tiles of 16; each B fragment
  // is reused across the two 16-row M-subtiles (halves L2 weight traffic).
  auto gemm = [&](const unsigned short* __restrict__ W,
                  const float* __restrict__ bias,
                  unsigned short* lds_out, float* g_out) {
    for (int j = 0; j < 4; ++j) {
      const int ncol = wave * NW + j * 16 + mcol;   // this lane's output column
      v8f acc0 = {0.f, 0.f, 0.f, 0.f, 0.f, 0.f, 0.f, 0.f};
      v8f acc1 = {0.f, 0.f, 0.f, 0.f, 0.f, 0.f, 0.f, 0.f};
      const unsigned short* ap0 = xs + mcol * F + kh * 8;            // rows 0..15
      const unsigned short* ap1 = xs + (16 + mcol) * F + kh * 8;     // rows 16..31
      const unsigned short* bp0 = W + (size_t)ncol * F + kh * 8;     // global B
      if (j < 3) __builtin_prefetch(bp0 + 16 * F, 0, 0);  // warm next N-tile row
      for (int kb = 0; kb < F; kb += 32) {
        union { v16bf v; uint4 q[2]; } A0, A1, B;
        B.q[0]  = *(const uint4*)(bp0 + kb);        // K = kb + kh*8 + [0..7]
        B.q[1]  = *(const uint4*)(bp0 + kb + 16);   // K = kb + 16 + kh*8 + [0..7]
        A0.q[0] = *(const uint4*)(ap0 + kb);
        A0.q[1] = *(const uint4*)(ap0 + kb + 16);
        A1.q[0] = *(const uint4*)(ap1 + kb);
        A1.q[1] = *(const uint4*)(ap1 + kb + 16);
        acc0 = __builtin_amdgcn_wmma_f32_16x16x32_bf16(
            false, A0.v, false, B.v, (short)0, acc0, false, false);
        acc1 = __builtin_amdgcn_wmma_f32_16x16x32_bf16(
            false, A1.v, false, B.v, (short)0, acc1, false, false);
      }
      const float bb = bias[ncol];
      if (lds_out) {
        for (int r = 0; r < 8; ++r) {
          const int row = r + kh * 8;              // C layout: VGPR r, lane-half
          lds_out[row * F + ncol]        = f2bf(acc0[r] + bb);
          lds_out[(16 + row) * F + ncol] = f2bf(acc1[r] + bb);
        }
      } else {
        for (int r = 0; r < 8; ++r) {
          const int row = r + kh * 8;
          g_out[(size_t)(t0 + row) * F + ncol]      = acc0[r] + bb;
          g_out[(size_t)(t0 + 16 + row) * F + ncol] = acc1[r] + bb;
        }
      }
    }
  };

  // ---- Q/K/V projections -------------------------------------------------
  load_tile(Qin); __syncthreads();
  gemm(Wqb, bq, qs, nullptr); __syncthreads();
  load_tile(Kin); __syncthreads();
  gemm(Wkb, bk, ks, nullptr); __syncthreads();
  load_tile(Vin); __syncthreads();
  gemm(Wvb, bv, vs, nullptr); __syncthreads();

  // ---- per-token 16x16 head-mixing attention (tiny; VALU) ----------------
  // 512 threads = 32 tokens x 16 head-rows: thread owns (token l, head-row h).
  {
    const int l = tid >> 4;
    const int h = tid & 15;
    const unsigned short* qrow = qs + l * F + h * HDIM;
    float attn[NHEAD];
    float mx = -3.4e38f;
    for (int e = 0; e < NHEAD; ++e) {
      const unsigned short* krow = ks + l * F + e * HDIM;
      float s = 0.f;
      for (int d = 0; d < HDIM; ++d) s += bf2f(qrow[d]) * bf2f(krow[d]);
      s *= 0.125f;                                  // 1/sqrt(HD=64)
      attn[e] = s;
      mx = fmaxf(mx, s);
    }
    float sum = 0.f;
    for (int e = 0; e < NHEAD; ++e) { attn[e] = __expf(attn[e] - mx); sum += attn[e]; }
    const float inv = 1.f / sum;
    // ctx[l, h, :] = softmax(attn) @ v[l, :, :]  -> reuse xs as ctx buffer
    for (int d = 0; d < HDIM; ++d) {
      float c = 0.f;
      for (int e = 0; e < NHEAD; ++e) c += attn[e] * bf2f(vs[l * F + e * HDIM + d]);
      xs[l * F + h * HDIM + d] = f2bf(c * inv);
    }
  }
  __syncthreads();

  // ---- output projection straight to global fp32 -------------------------
  gemm(Wob, bo, nullptr, out);
}

// ---------------------------------------------------------------------------
// Host launcher
// ---------------------------------------------------------------------------
extern "C" void kernel_launch(void* const* d_in, const int* in_sizes, int n_in,
                              void* d_out, int out_size, void* d_ws, size_t ws_size,
                              hipStream_t stream) {
  const float* Q  = (const float*)d_in[0];
  const float* K  = (const float*)d_in[1];
  const float* V  = (const float*)d_in[2];
  const float* Wq = (const float*)d_in[3];
  const float* bq = (const float*)d_in[4];
  const float* Wk = (const float*)d_in[5];
  const float* bk = (const float*)d_in[6];
  const float* Wv = (const float*)d_in[7];
  const float* bv = (const float*)d_in[8];
  const float* Wo = (const float*)d_in[9];
  const float* bo = (const float*)d_in[10];
  float* out = (float*)d_out;

  const int WN = F * F;                          // 1M elements per weight matrix
  unsigned short* wq = (unsigned short*)d_ws;    // 2 MB each, 8 MB total
  unsigned short* wk = wq + WN;
  unsigned short* wv = wk + WN;
  unsigned short* wo = wv + WN;

  dim3 cgrid(WN / 4 / 256);
  cvt_w_kernel<<<cgrid, 256, 0, stream>>>(Wq, wq, WN);
  cvt_w_kernel<<<cgrid, 256, 0, stream>>>(Wk, wk, WN);
  cvt_w_kernel<<<cgrid, 256, 0, stream>>>(Wv, wv, WN);
  cvt_w_kernel<<<cgrid, 256, 0, stream>>>(Wo, wo, WN);

  const size_t smem_bytes = (size_t)4 * TM * F * sizeof(unsigned short); // 256 KB
  (void)hipFuncSetAttribute((const void*)mha_fused_kernel,
                            hipFuncAttributeMaxDynamicSharedMemorySize,
                            (int)smem_bytes);
  mha_fused_kernel<<<NTOK / TM, THREADS, smem_bytes, stream>>>(
      Q, K, V, wq, bq, wk, bk, wv, bv, wo, bo, out);
}
